// InjectedEncoder_72911364816949
// MI455X (gfx1250) — matrix operand order repaired and checked by reading, not compiled, not run on hardware
//
#include <hip/hip_runtime.h>
#include <hip/hip_bf16.h>

// ---------------------------------------------------------------------------
// CDNA5 / gfx1250 implementation of InjectedEncoder forward.
// All matmuls (hypernet weight generation + per-sample 1x1 convs) run through
// v_wmma_f32_16x16x32_bf16 (bf16 operands staged via LDS, f32 accumulate).
// ---------------------------------------------------------------------------

typedef __bf16 bf16_t;
typedef __attribute__((ext_vector_type(16))) __bf16 v16bf;
typedef __attribute__((ext_vector_type(8)))  __bf16 v8bf;
typedef __attribute__((ext_vector_type(4)))  __bf16 v4bf;
typedef __attribute__((ext_vector_type(8)))  float  v8f;
typedef __attribute__((ext_vector_type(4)))  float  v4f;

#define TBM 64
#define TBN 128
#define TBK 32
// A tile: [TBM][LDA_S] bf16, 96-byte rows -> both 8-elem fragment runs are
// 16B aligned (ds_load_b128).
#define LDA_S 48
// B tile stored TRANSPOSED: [TBN][LDBT] bf16, 80-byte rows -> each
// 16-elem fragment (k = 8*half + e, then 16 + 8*half + e) is two aligned ds_load_b128.
#define LDBT 40

// ---------------------------------------------------------------------------
// WMMA GEMM:  C[b] = act( A[b](MxK) * B[b](KxN) + bias ) (+ C if accum)
//   biasMode: 0 none, 1 per-row (bias[M]), 2 per-col (bias[N])
//   actMode : 0 none, 1 leaky-relu(0.2)
//   strides sA/sB/sC/sBias give per-batch element strides (0 = shared).
// 8 waves, wave grid 2x4, each wave computes a 32x32 tile as 2x2 WMMA tiles.
// ---------------------------------------------------------------------------
__global__ __launch_bounds__(256) void wmma_gemm_kernel(
    const float* __restrict__ A, long long sA, int lda,
    const float* __restrict__ B, long long sB, int ldb,
    float* __restrict__ C, long long sC, int ldc,
    const float* __restrict__ bias, long long sBias, int biasMode,
    int M, int N, int K, int actMode, int accum)
{
  __shared__ bf16_t As[TBM * LDA_S];   // 6 KB
  __shared__ bf16_t BsT[TBN * LDBT];   // 10 KB

  const int tid  = threadIdx.x;
  const int lane = tid & 31;
  const int wav  = tid >> 5;          // 0..7
  const int wm   = (wav & 1) * 32;    // wave M offset in block tile
  const int wn   = (wav >> 1) * 32;   // wave N offset in block tile
  const int bm   = blockIdx.y * TBM;
  const int bn   = blockIdx.x * TBN;
  const int bz   = blockIdx.z;

  const float* Ab = A + (long long)bz * sA;
  const float* Bb = B + (long long)bz * sB;
  float*       Cb = C + (long long)bz * sC;

  v8f acc[2][2];
#pragma unroll
  for (int i = 0; i < 2; ++i)
#pragma unroll
    for (int j = 0; j < 2; ++j)
#pragma unroll
      for (int r = 0; r < 8; ++r) acc[i][j][r] = 0.f;

  const int half = lane >> 4;   // 0/1
  const int l16  = lane & 15;

  const bool aInteriorM = (bm + TBM <= M) && ((lda & 3) == 0);
  const bool bInteriorN = (bn + TBN <= N) && ((ldb & 3) == 0);

  for (int k0 = 0; k0 < K; k0 += TBK) {
    const bool kFull = (k0 + TBK <= K);

    // ---- stage A tile [TBM x TBK] as bf16 -------------------------------
    if (aInteriorM && kFull) {
      const float* Abase = Ab + (long long)bm * lda + k0;
#pragma unroll
      for (int j = 0; j < 2; ++j) {
        int f  = tid + j * 256;          // 0..511 float4s
        int r  = f >> 3;                 // row 0..63
        int k4 = (f & 7) << 2;           // 0,4,...,28
        v4f v = *(const v4f*)(Abase + (long long)r * lda + k4);
        v4bf t;
        t[0] = (bf16_t)v[0]; t[1] = (bf16_t)v[1];
        t[2] = (bf16_t)v[2]; t[3] = (bf16_t)v[3];
        *(v4bf*)&As[r * LDA_S + k4] = t;
      }
    } else {
      for (int i = tid; i < TBM * TBK; i += 256) {
        int r = i >> 5, kk = i & 31;
        int gm = bm + r, gk = k0 + kk;
        int gmc = gm < M ? gm : M - 1;
        int gkc = gk < K ? gk : K - 1;
        float v = Ab[(long long)gmc * lda + gkc];   // clamped, branch-free
        As[r * LDA_S + kk] = (bf16_t)((gm < M && gk < K) ? v : 0.f);
      }
    }

    // ---- stage B tile [TBK x TBN] TRANSPOSED to [TBN][TBK] --------------
    if (bInteriorN && kFull) {
      const float* Bbase = Bb + (long long)k0 * ldb + bn;
      int n4 = tid & 31;                 // float4 column 0..31
      int kb = (tid >> 5) << 2;          // k row group 0,4,...,28
      v4f r0 = *(const v4f*)(Bbase + (long long)(kb + 0) * ldb + n4 * 4);
      v4f r1 = *(const v4f*)(Bbase + (long long)(kb + 1) * ldb + n4 * 4);
      v4f r2 = *(const v4f*)(Bbase + (long long)(kb + 2) * ldb + n4 * 4);
      v4f r3 = *(const v4f*)(Bbase + (long long)(kb + 3) * ldb + n4 * 4);
#pragma unroll
      for (int j = 0; j < 4; ++j) {      // 4x4 register transpose
        v4bf t;
        t[0] = (bf16_t)r0[j]; t[1] = (bf16_t)r1[j];
        t[2] = (bf16_t)r2[j]; t[3] = (bf16_t)r3[j];
        *(v4bf*)&BsT[(n4 * 4 + j) * LDBT + kb] = t;
      }
    } else {
      for (int i = tid; i < TBK * TBN; i += 256) {
        int kk = i >> 7, n = i & 127;
        int gk = k0 + kk, gn = bn + n;
        int gkc = gk < K ? gk : K - 1;
        int gnc = gn < N ? gn : N - 1;
        float v = Bb[(long long)gkc * ldb + gnc];
        BsT[n * LDBT + kk] = (bf16_t)((gk < K && gn < N) ? v : 0.f);
      }
    }

    // prefetch next K-step tiles into cache (global_prefetch_b8)
    if (k0 + TBK < K) {
      __builtin_prefetch(Ab + (long long)(bm + (tid & 63)) * lda + k0 + TBK, 0, 0);
      __builtin_prefetch(Bb + (long long)(k0 + TBK + (tid & 31)) * ldb + bn + ((tid >> 5) << 4), 0, 0);
    }
    __syncthreads();

    // ---- fragments: 2x ds_load_b128 each (ISA 7.12.2 16-bit layouts) ----
    v16bf af[2], bfr[2];
#pragma unroll
    for (int mi = 0; mi < 2; ++mi) {
      const bf16_t* p = &As[(wm + mi * 16 + l16) * LDA_S + 8 * half];
      v8bf lo = *(const v8bf*)p;          // k = 8*half + 0..7
      v8bf hi = *(const v8bf*)(p + 16);   // k = 16 + 8*half + 0..7
      af[mi] = __builtin_shufflevector(lo, hi, 0, 1, 2, 3, 4, 5, 6, 7,
                                       8, 9, 10, 11, 12, 13, 14, 15);
    }
#pragma unroll
    for (int ni = 0; ni < 2; ++ni) {
      const bf16_t* p = &BsT[(wn + ni * 16 + l16) * LDBT + 8 * half];
      v8bf lo = *(const v8bf*)p;          // k = 8*half + 0..7
      v8bf hi = *(const v8bf*)(p + 16);   // k = 16 + 8*half + 0..7
      bfr[ni] = __builtin_shufflevector(lo, hi, 0, 1, 2, 3, 4, 5, 6, 7,
                                        8, 9, 10, 11, 12, 13, 14, 15);
    }
#pragma unroll
    for (int mi = 0; mi < 2; ++mi)
#pragma unroll
      for (int ni = 0; ni < 2; ++ni)
        acc[mi][ni] = __builtin_amdgcn_wmma_f32_16x16x32_bf16(
            false, af[mi], false, bfr[ni], (short)0, acc[mi][ni], false, false);
    __syncthreads();
  }

  // ---- epilogue: C/D layout — VGPR r: row = r + 8*half, col = lane&15 ----
  const bool interior = (bm + TBM <= M) && (bn + TBN <= N);
#pragma unroll
  for (int mi = 0; mi < 2; ++mi)
#pragma unroll
    for (int ni = 0; ni < 2; ++ni) {
      int gm0 = bm + wm + mi * 16 + 8 * half;
      int gn  = bn + wn + ni * 16 + l16;
#pragma unroll
      for (int r = 0; r < 8; ++r) {
        int gm = gm0 + r;
        if (interior || (gm < M && gn < N)) {
          float v = acc[mi][ni][r];
          if (biasMode == 1)      v += bias[(long long)bz * sBias + gm];
          else if (biasMode == 2) v += bias[(long long)bz * sBias + gn];
          if (actMode == 1) v = v > 0.f ? v : 0.2f * v;
          long long off = (long long)gm * ldc + gn;
          if (accum) v += Cb[off];
          Cb[off] = v;
        }
      }
    }
}

// ---------------------------------------------------------------------------
// Elementwise / reduction kernels
// ---------------------------------------------------------------------------
__global__ void posenc_kernel(float* __restrict__ cw, int S, int F) {
  int idx = blockIdx.x * blockDim.x + threadIdx.x;
  int total = 4 * F * S * S;
  if (idx >= total) return;
  int xx = idx % S;
  int yy = (idx / S) % S;
  int ch = idx / (S * S);
  int k = (ch < 2 * F) ? ch : ch - 2 * F;
  int p = (ch < 2 * F) ? yy : xx;
  float w = exp2f((float)(k >> 1)) * 3.14159265358979323846f;
  float a = w * ((p + 0.5f) / (float)S);
  cw[idx] = (k & 1) ? cosf(a) : sinf(a);
}

__global__ void wave_b_kernel(const float* __restrict__ surf,
                              float* __restrict__ wb, int HW) {
  long long idx = (long long)blockIdx.x * blockDim.x + threadIdx.x;
  long long total = (long long)8 * 64 * HW;
  if (idx >= total) return;
  int hw = (int)(idx % HW);
  int c  = (int)((idx / HW) % 64);
  int b  = (int)(idx / ((long long)HW * 64));
  float s = surf[((long long)b * 32 + (c & 31)) * HW + hw];
  const float Ec = 2.718281828459045f;
  wb[idx] = (c < 32) ? Ec * cosf(s) : Ec * sinf(s);
}

__global__ void sobel_kernel(const float* __restrict__ x,
                             float* __restrict__ g, int S) {
  long long idx = (long long)blockIdx.x * blockDim.x + threadIdx.x;
  long long total = (long long)8 * 64 * S * S;
  if (idx >= total) return;
  int xx = (int)(idx % S);
  int yy = (int)((idx / S) % S);
  int c  = (int)((idx / ((long long)S * S)) % 64);
  int b  = (int)(idx / ((long long)S * S * 64));
  const float* p = x + ((long long)b * 32 + (c & 31)) * S * S;
  float v;
  if (c < 32) {  // d/dx, zero-padded SAME
    float r = (xx + 1 < S) ? p[yy * S + xx + 1] : 0.f;
    float l = (xx - 1 >= 0) ? p[yy * S + xx - 1] : 0.f;
    v = r - l;
  } else {       // d/dy
    float d = (yy + 1 < S) ? p[(yy + 1) * S + xx] : 0.f;
    float u = (yy - 1 >= 0) ? p[(yy - 1) * S + xx] : 0.f;
    v = d - u;
  }
  g[idx] = v;
}

__global__ void chan_mean_kernel(const float* __restrict__ x,
                                 float* __restrict__ m, int HW) {
  __shared__ float sd[256];
  const float* p = x + (long long)blockIdx.x * HW;
  float s = 0.f;
  for (int i = threadIdx.x; i < HW; i += 256) s += p[i];
  sd[threadIdx.x] = s;
  __syncthreads();
  for (int o = 128; o > 0; o >>= 1) {
    if ((int)threadIdx.x < o) sd[threadIdx.x] += sd[threadIdx.x + o];
    __syncthreads();
  }
  if (threadIdx.x == 0) m[blockIdx.x] = sd[0] / (float)HW;
}

// wave_c mean: dyna_conv then spatial mean == matvec on channel means (linear)
__global__ void wavec_matvec_kernel(const float* __restrict__ w,
                                    const float* __restrict__ bias,
                                    const float* __restrict__ om,
                                    float* __restrict__ wc, int fout) {
  int idx = blockIdx.x * blockDim.x + threadIdx.x;
  if (idx >= 8 * fout) return;
  int b = idx / fout, o = idx % fout;
  const float* wp = w + ((long long)b * fout + o) * 32;
  const float* op = om + b * 32;
  float s = bias[b * fout + o];
  for (int i = 0; i < 32; ++i) s += wp[i] * op[i];
  wc[idx] = s;
}

__global__ void wave_in_kernel(const float* __restrict__ wc,
                               const float* __restrict__ cellw,
                               const float* __restrict__ wb,
                               float* __restrict__ wi, int cw, int HW) {
  int C = cw + 64;
  long long idx = (long long)blockIdx.x * blockDim.x + threadIdx.x;
  long long total = (long long)8 * C * HW;
  if (idx >= total) return;
  int hw = (int)(idx % HW);
  int c  = (int)((idx / HW) % C);
  int b  = (int)(idx / ((long long)HW * C));
  float v;
  if (c < cw) v = wc[b * cw + c] * cellw[(long long)c * HW + hw];
  else        v = wb[((long long)b * 64 + (c - cw)) * HW + hw];
  wi[idx] = v;
}

__global__ void inorm_kernel(float* __restrict__ x, int HW) {
  __shared__ float s1[256], s2[256];
  float* p = x + (long long)blockIdx.x * HW;
  float a = 0.f, q = 0.f;
  for (int i = threadIdx.x; i < HW; i += 256) {
    float v = p[i];
    a += v;
    q += v * v;
  }
  s1[threadIdx.x] = a;
  s2[threadIdx.x] = q;
  __syncthreads();
  for (int o = 128; o > 0; o >>= 1) {
    if ((int)threadIdx.x < o) {
      s1[threadIdx.x] += s1[threadIdx.x + o];
      s2[threadIdx.x] += s2[threadIdx.x + o];
    }
    __syncthreads();
  }
  float mean = s1[0] / (float)HW;
  float var  = s2[0] / (float)HW - mean * mean;
  float inv  = rsqrtf(var + 1e-5f);
  for (int i = threadIdx.x; i < HW; i += 256) p[i] = (p[i] - mean) * inv;
}

__global__ void axpy_leak_kernel(float* __restrict__ out,
                                 const float* __restrict__ on,
                                 const float* __restrict__ leakp, long long n) {
  long long i = (long long)blockIdx.x * blockDim.x + threadIdx.x;
  if (i >= n) return;
  float lf = fminf(fmaxf(leakp[0], 0.001f), 1000.f);
  out[i] += lf * on[i];
}

// 3x3 gaussian (sigma=1) + 2x2 average (bilinear 0.5x) fused
__global__ void gauss_down2_kernel(const float* __restrict__ x,
                                   float* __restrict__ y, int S) {
  int So = S >> 1;
  long long idx = (long long)blockIdx.x * blockDim.x + threadIdx.x;
  long long total = (long long)8 * 32 * So * So;
  if (idx >= total) return;
  int xo = (int)(idx % So);
  int yo = (int)((idx / So) % So);
  int c  = (int)((idx / ((long long)So * So)) % 32);
  int b  = (int)(idx / ((long long)So * So * 32));
  const float* p = x + ((long long)b * 32 + c) * S * S;
  const float w0 = 0.60653065971263342f;  // exp(-0.5)
  const float s1 = 1.f + 2.f * w0;
  const float inv = 1.f / (s1 * s1);
  float acc = 0.f;
  for (int dy2 = 0; dy2 < 2; ++dy2)
    for (int dx2 = 0; dx2 < 2; ++dx2) {
      int yi = 2 * yo + dy2, xi = 2 * xo + dx2;
      float g = 0.f;
      for (int dy = -1; dy <= 1; ++dy)
        for (int dx = -1; dx <= 1; ++dx) {
          int yy = yi + dy, xx = xi + dx;
          if (yy >= 0 && yy < S && xx >= 0 && xx < S) {
            float wgt = ((dy == 0) ? 1.f : w0) * ((dx == 0) ? 1.f : w0);
            g += wgt * p[yy * S + xx];
          }
        }
      acc += g * inv;
    }
  y[idx] = acc * 0.25f;
}

// pooled-state assembly from cs[B,224,16,16] -> state[B,1345]
__global__ void state_kernel(const float* __restrict__ cs,
                             float* __restrict__ state) {
  int b = blockIdx.x;
  const float* base = cs + (long long)b * 224 * 256;
  for (int o = threadIdx.x; o < 1345; o += blockDim.x) {
    float v = 0.f;
    if (o < 32) {                               // f_: mean HW
      const float* p = base + o * 256;
      for (int i = 0; i < 256; ++i) v += p[i];
      v *= (1.f / 256.f);
    } else if (o < 48) {                        // h_: mean (C,W) per row
      int yy = o - 32;
      for (int c = 0; c < 32; ++c)
        for (int xx = 0; xx < 16; ++xx) v += base[(32 + c) * 256 + yy * 16 + xx];
      v *= (1.f / 512.f);
    } else if (o < 64) {                        // w_: mean (C,H) per col
      int xx = o - 48;
      for (int c = 0; c < 32; ++c)
        for (int yy = 0; yy < 16; ++yy) v += base[(64 + c) * 256 + yy * 16 + xx];
      v *= (1.f / 512.f);
    } else if (o < 576) {                       // fh: mean W, (c,y)
      int t = o - 64, c = t / 16, yy = t % 16;
      for (int xx = 0; xx < 16; ++xx) v += base[(96 + c) * 256 + yy * 16 + xx];
      v *= (1.f / 16.f);
    } else if (o < 1088) {                      // fw: mean H, (c,x)
      int t = o - 576, c = t / 16, xx = t % 16;
      for (int yy = 0; yy < 16; ++yy) v += base[(128 + c) * 256 + yy * 16 + xx];
      v *= (1.f / 16.f);
    } else if (o < 1344) {                      // hw: mean C, (y,x)
      int t = o - 1088, yy = t / 16, xx = t % 16;
      for (int c = 0; c < 32; ++c) v += base[(160 + c) * 256 + yy * 16 + xx];
      v *= (1.f / 32.f);
    } else {                                    // g_: mean all
      for (int c = 0; c < 32; ++c)
        for (int i = 0; i < 256; ++i) v += base[(192 + c) * 256 + i];
      v *= (1.f / 8192.f);
    }
    state[b * 1345 + o] = v;
  }
}

// ---------------------------------------------------------------------------
// Host orchestration
// ---------------------------------------------------------------------------
static inline unsigned cdiv(long long a, int b) {
  return (unsigned)((a + b - 1) / b);
}

static void launch_gemm(hipStream_t s,
                        const float* A, long long sA, int lda,
                        const float* B, long long sB, int ldb,
                        float* C, long long sC, int ldc,
                        const float* bias, long long sBias, int biasMode,
                        int M, int N, int K, int act, int accum, int batch) {
  dim3 grid((N + TBN - 1) / TBN, (M + TBM - 1) / TBM, batch);
  wmma_gemm_kernel<<<grid, dim3(256), 0, s>>>(A, sA, lda, B, sB, ldb, C, sC,
                                              ldc, bias, sBias, biasMode, M, N,
                                              K, act, accum);
}

extern "C" void kernel_launch(void* const* d_in, const int* in_sizes, int n_in,
                              void* d_out, int out_size, void* d_ws,
                              size_t ws_size, hipStream_t stream) {
  (void)in_sizes; (void)n_in; (void)out_size; (void)ws_size;

  const float* x   = (const float*)d_in[0];
  const float* inj = (const float*)d_in[1];
  auto P = [&](int i) -> const float* { return (const float*)d_in[i]; };

  // jax pytree (sorted-key) flattening of params, leaves start at index 2:
  //   bias_trans_l: 2+4l  {bb,bw,wb,ww}
  //   cell_to_cell_l: 18+14l {conv1.bb,bw,wb,ww, conv2.bb,bw,wb,ww,
  //                           lat_b, lat_w, skip.bb,bw,wb,ww}
  //   cell_to_wave_l: 74+4l ; conv_img_b=90 conv_img_w=91
  //   frac_lat_exp_l: 92+2l {b,w} ; grad_to_cell_l: 100+4l
  //   leak_factor=116 ; out_conv_b=117 out_conv_w=118
  //   out_to_lat_b=119 out_to_lat_w=120 ; wave_bias_l=121+l
  //   wave_to_cell_l: 125+4l
  const int HW0 = 128 * 128;

  float* W = (float*)d_ws;
  size_t off = 0;
  auto A_ = [&](size_t n) { size_t r = off; off += (n + 63) & ~(size_t)63; return r; };

  size_t oOUT    = A_((size_t)8 * 32 * HW0);
  size_t oOUTNEW = A_((size_t)8 * 32 * HW0);
  size_t oWAVEB  = A_((size_t)8 * 64 * HW0);
  size_t oGRADS  = A_((size_t)8 * 64 * HW0);   // also reused as dyna_res hidden
  size_t oCAT    = A_((size_t)8 * 128 * HW0);
  size_t oWAVEIN = A_((size_t)8 * 96 * HW0);
  size_t oCELLW  = A_((size_t)28 * HW0);
  size_t oLATNEW = A_(8 * 512);
  size_t oLAT2   = A_(8 * 1024);
  size_t oOM     = A_(8 * 32);
  size_t oWC     = A_(8 * 32);
  size_t oWBT  = A_(8 * 16384), oBBT  = A_(8 * 32);
  size_t oWC2W = A_(8 * 32 * 28), oBC2W = A_(8 * 28);
  size_t oWW2C = A_((size_t)8 * 92 * 64), oBW2C = A_(8 * 64);
  size_t oWG2C = A_(8 * 64 * 64), oBG2C = A_(8 * 64);
  size_t oWCV1 = A_((size_t)8 * 128 * 64), oBCV1 = A_(8 * 64);
  size_t oWCV2 = A_(8 * 64 * 32), oBCV2 = A_(8 * 32);
  size_t oWSKP = A_((size_t)8 * 128 * 32), oBSKP = A_(8 * 32);
  size_t oCS   = A_((size_t)8 * 224 * 256);
  size_t oSTATE = A_(8 * 1345);

  float* out    = W + oOUT;
  float* outNew = W + oOUTNEW;

  int S = 128;
  float* emb = (float*)d_out + 8 * 512;  // lat goes first, then out_embs

  // out = conv_img(x): GEMM M=32,K=3 (zero-padded to 32 in LDS), shared weights
  launch_gemm(stream, P(91), 0, 3, x, (long long)3 * HW0, HW0,
              out, (long long)32 * HW0, HW0, P(90), 0, 1, 32, HW0, 3, 0, 0, 8);
  hipMemcpyAsync(emb, out, (size_t)8 * 32 * HW0 * sizeof(float),
                 hipMemcpyDeviceToDevice, stream);
  emb += (size_t)8 * 32 * HW0;

  for (int c = 0; c < 16; ++c) {
    int l  = c >> 2;
    int HW = S * S;
    int F  = 7 - l;
    int cw = 4 * F;
    int bt = 2 + 4 * l, c2c = 18 + 14 * l, c2w = 74 + 4 * l, fle = 92 + 2 * l;
    int g2c = 100 + 4 * l, wbs = 121 + l, w2c = 125 + 4 * l;

    if ((c & 3) == 0) {
      // positional encoding for this resolution
      posenc_kernel<<<cdiv((long long)4 * F * HW, 256), 256, 0, stream>>>(
          W + oCELLW, S, F);
      // lat_new = inj @ w + b ; lat2 = lat_new @ lat_w + lat_b
      launch_gemm(stream, inj, 0, 512, P(fle + 1), 0, 512, W + oLATNEW, 0, 512,
                  P(fle + 0), 0, 2, 8, 512, 512, 0, 0, 1);
      launch_gemm(stream, W + oLATNEW, 0, 512, P(c2c + 9), 0, 1024, W + oLAT2,
                  0, 1024, P(c2c + 8), 0, 2, 8, 1024, 512, 0, 0, 1);
      // hypernet weight/bias generation (per layer, reused across 4 calls)
      launch_gemm(stream, W + oLATNEW, 0, 512, P(bt + 3), 0, 16384, W + oWBT, 0,
                  16384, P(bt + 2), 0, 2, 8, 16384, 512, 0, 0, 1);
      launch_gemm(stream, W + oLATNEW, 0, 512, P(bt + 1), 0, 32, W + oBBT, 0,
                  32, P(bt + 0), 0, 2, 8, 32, 512, 0, 0, 1);
      launch_gemm(stream, W + oLATNEW, 0, 512, P(c2w + 3), 0, 32 * cw,
                  W + oWC2W, 0, 32 * cw, P(c2w + 2), 0, 2, 8, 32 * cw, 512, 0, 0, 1);
      launch_gemm(stream, W + oLATNEW, 0, 512, P(c2w + 1), 0, cw, W + oBC2W, 0,
                  cw, P(c2w + 0), 0, 2, 8, cw, 512, 0, 0, 1);
      int fw2c = (cw + 64) * 64;
      launch_gemm(stream, W + oLATNEW, 0, 512, P(w2c + 3), 0, fw2c, W + oWW2C,
                  0, fw2c, P(w2c + 2), 0, 2, 8, fw2c, 512, 0, 0, 1);
      launch_gemm(stream, W + oLATNEW, 0, 512, P(w2c + 1), 0, 64, W + oBW2C, 0,
                  64, P(w2c + 0), 0, 2, 8, 64, 512, 0, 0, 1);
      launch_gemm(stream, W + oLATNEW, 0, 512, P(g2c + 3), 0, 4096, W + oWG2C,
                  0, 4096, P(g2c + 2), 0, 2, 8, 4096, 512, 0, 0, 1);
      launch_gemm(stream, W + oLATNEW, 0, 512, P(g2c + 1), 0, 64, W + oBG2C, 0,
                  64, P(g2c + 0), 0, 2, 8, 64, 512, 0, 0, 1);
      launch_gemm(stream, W + oLAT2, 0, 1024, P(c2c + 3), 0, 8192, W + oWCV1, 0,
                  8192, P(c2c + 2), 0, 2, 8, 8192, 1024, 0, 0, 1);
      launch_gemm(stream, W + oLAT2, 0, 1024, P(c2c + 1), 0, 64, W + oBCV1, 0,
                  64, P(c2c + 0), 0, 2, 8, 64, 1024, 0, 0, 1);
      launch_gemm(stream, W + oLAT2, 0, 1024, P(c2c + 7), 0, 2048, W + oWCV2, 0,
                  2048, P(c2c + 6), 0, 2, 8, 2048, 1024, 0, 0, 1);
      launch_gemm(stream, W + oLAT2, 0, 1024, P(c2c + 5), 0, 32, W + oBCV2, 0,
                  32, P(c2c + 4), 0, 2, 8, 32, 1024, 0, 0, 1);
      launch_gemm(stream, W + oLAT2, 0, 1024, P(c2c + 13), 0, 4096, W + oWSKP,
                  0, 4096, P(c2c + 12), 0, 2, 8, 4096, 1024, 0, 0, 1);
      launch_gemm(stream, W + oLAT2, 0, 1024, P(c2c + 11), 0, 32, W + oBSKP, 0,
                  32, P(c2c + 10), 0, 2, 8, 32, 1024, 0, 0, 1);
      // surf = bias_trans(wave_bias)  (surf staged in outNew), then wave_b
      launch_gemm(stream, W + oWBT, 16384, 512, P(wbs), 0, HW, outNew,
                  (long long)32 * HW, HW, W + oBBT, 32, 1, 32, HW, 512, 0, 0, 8);
      wave_b_kernel<<<cdiv((long long)8 * 64 * HW, 256), 256, 0, stream>>>(
          outNew, W + oWAVEB, HW);
    }

    // wave_c: channel means of `out`, then small per-sample matvec (linear)
    chan_mean_kernel<<<8 * 32, 256, 0, stream>>>(out, W + oOM, HW);
    wavec_matvec_kernel<<<cdiv(8 * cw, 64), 64, 0, stream>>>(
        W + oWC2W, W + oBC2W, W + oOM, W + oWC, cw);
    wave_in_kernel<<<cdiv((long long)8 * (cw + 64) * HW, 256), 256, 0, stream>>>(
        W + oWC, W + oCELLW, W + oWAVEB, W + oWAVEIN, cw, HW);
    // wave_enc -> CAT[64:128]
    launch_gemm(stream, W + oWW2C, (long long)(cw + 64) * 64, cw + 64,
                W + oWAVEIN, (long long)(cw + 64) * HW, HW,
                W + oCAT + (long long)64 * HW, (long long)128 * HW, HW,
                W + oBW2C, 64, 1, 64, HW, cw + 64, 0, 0, 8);
    // grads_enc -> CAT[0:64]
    sobel_kernel<<<cdiv((long long)8 * 64 * HW, 256), 256, 0, stream>>>(
        out, W + oGRADS, S);
    launch_gemm(stream, W + oWG2C, 4096, 64, W + oGRADS, (long long)64 * HW, HW,
                W + oCAT, (long long)128 * HW, HW, W + oBG2C, 64, 1, 64, HW, 64,
                0, 0, 8);
    // per-channel instance norm over both halves of CAT
    inorm_kernel<<<8 * 128, 256, 0, stream>>>(W + oCAT, HW);
    // dyna_res: out_new = skip(cat) + conv2(leaky(conv1(cat)))
    launch_gemm(stream, W + oWSKP, 4096, 128, W + oCAT, (long long)128 * HW, HW,
                outNew, (long long)32 * HW, HW, W + oBSKP, 32, 1, 32, HW, 128,
                0, 0, 8);
    launch_gemm(stream, W + oWCV1, 8192, 128, W + oCAT, (long long)128 * HW, HW,
                W + oGRADS, (long long)64 * HW, HW, W + oBCV1, 64, 1, 64, HW,
                128, 1 /*leaky*/, 0, 8);
    launch_gemm(stream, W + oWCV2, 2048, 64, W + oGRADS, (long long)64 * HW, HW,
                outNew, (long long)32 * HW, HW, W + oBCV2, 32, 1, 32, HW, 64, 0,
                1 /*accum*/, 8);
    // out += leak * out_new
    axpy_leak_kernel<<<cdiv((long long)8 * 32 * HW, 256), 256, 0, stream>>>(
        out, outNew, P(116), (long long)8 * 32 * HW);

    if (c < 15 && (c & 3) == 3) {  // gaussian smooth + 0.5x downsample
      gauss_down2_kernel<<<cdiv((long long)8 * 32 * (S / 2) * (S / 2), 256),
                           256, 0, stream>>>(out, outNew, S);
      float* t = out; out = outNew; outNew = t;
      S >>= 1;
    }
    hipMemcpyAsync(emb, out, (size_t)8 * 32 * S * S * sizeof(float),
                   hipMemcpyDeviceToDevice, stream);
    emb += (size_t)8 * 32 * S * S;
  }

  // cs = out_conv(out) at S=16 ; pooled state ; final lat projection
  launch_gemm(stream, P(118), 0, 32, out, (long long)32 * 256, 256, W + oCS,
              (long long)224 * 256, 256, P(117), 0, 1, 224, 256, 32, 0, 0, 8);
  state_kernel<<<8, 256, 0, stream>>>(W + oCS, W + oSTATE);
  launch_gemm(stream, W + oSTATE, 0, 1345, P(120), 0, 512, (float*)d_out, 0,
              512, P(119), 0, 2, 8, 512, 1345, 0, 0, 1);
}
